// multihead_attention_1211180777775
// MI455X (gfx1250) — matrix operand order repaired and checked by reading, not compiled, not run on hardware
//
#include <hip/hip_runtime.h>
#include <hip/hip_bf16.h>

// ---------------------------------------------------------------------------
// MHA forward, gfx1250 (MI455X), wave32, bf16 WMMA (f32 accumulate),
// async-to-LDS staging of shared operands (K/V tiles, weight tiles).
// B=8, S=1024, E=1024, H=16, Dh=64.
// ---------------------------------------------------------------------------

#define B_  8
#define S_  1024
#define E_  1024
#define H_  16
#define DH_ 64

typedef __attribute__((ext_vector_type(16))) __bf16 v16bf;
typedef __attribute__((ext_vector_type(8)))  __bf16 v8bf;
typedef __attribute__((ext_vector_type(8)))  float  v8f;

static __device__ __forceinline__ v8f wmma_bf16(v16bf a, v16bf b, v8f c) {
    return __builtin_amdgcn_wmma_f32_16x16x32_bf16(
        /*neg_a=*/false, a, /*neg_b=*/false, b,
        /*c_mod=*/(short)0, c, /*reuse_a=*/false, /*reuse_b=*/false);
}

// Async DMA: 16 bytes per lane, global -> LDS, tracked by ASYNCcnt.
// GVS mode: uniform SGPR base + per-lane 32-bit byte offset.
// LDS address = low 32 bits of the generic pointer (aperture offset).
static __device__ __forceinline__ void async_copy_b128(const __bf16* lds_dst,
                                                       const void* gbase,
                                                       unsigned goff_bytes) {
    unsigned lva = (unsigned)(size_t)lds_dst;
    asm volatile("global_load_async_to_lds_b128 %0, %1, %2"
                 :: "v"(lva), "v"(goff_bytes), "s"(gbase)
                 : "memory");
}
static __device__ __forceinline__ void wait_async_le2() {
    asm volatile("s_wait_asynccnt 0x2" ::: "memory");
}
static __device__ __forceinline__ void wait_async_0() {
    asm volatile("s_wait_asynccnt 0x0" ::: "memory");
}

// A-fragment (16x32, MxK) from row-major bf16 matrix (row r, K contiguous).
static __device__ __forceinline__ v16bf load_a_frag(const __bf16* __restrict__ base,
                                                    int m0, int k0, int ld) {
    int lane = threadIdx.x & 31;
    int r  = m0 + (lane & 15);
    int kb = k0 + ((lane >> 4) << 3);
    const __bf16* p = base + (size_t)r * ld + kb;
    v8bf lo = *reinterpret_cast<const v8bf*>(p);
    v8bf hi = *reinterpret_cast<const v8bf*>(p + 16);
    v16bf a;
#pragma unroll
    for (int i = 0; i < 8; ++i) { a[i] = lo[i]; a[i + 8] = hi[i]; }
    return a;
}

// B-fragment (32x16, KxN): element (k,n) at n*ld + k (K contiguous per column).
static __device__ __forceinline__ v16bf load_b_frag(const __bf16* base,
                                                    int n0, int k0, int ld) {
    int lane = threadIdx.x & 31;
    int n  = n0 + (lane & 15);
    int kb = k0 + ((lane >> 4) << 3);
    const __bf16* p = base + (size_t)n * ld + kb;
    v8bf lo = *reinterpret_cast<const v8bf*>(p);
    v8bf hi = *reinterpret_cast<const v8bf*>(p + 16);
    v16bf b;
#pragma unroll
    for (int i = 0; i < 8; ++i) { b[i] = lo[i]; b[i + 8] = hi[i]; }
    return b;
}

// ---------------------------------------------------------------------------
// f32 -> bf16 conversions
// ---------------------------------------------------------------------------
__global__ void cvt_x_bf16(const float* __restrict__ x, __bf16* __restrict__ xb, int n) {
    int i = blockIdx.x * blockDim.x + threadIdx.x;
    if (i < n) xb[i] = (__bf16)x[i];
}

// W [H,E,Dh] f32 -> Wt [H,Dh,E] bf16 (transpose so B-frags are K-contiguous)
__global__ void cvt_w_t_bf16(const float* __restrict__ W, __bf16* __restrict__ Wt) {
    int i = blockIdx.x * blockDim.x + threadIdx.x;     // over H*Dh*E
    if (i >= H_ * DH_ * E_) return;
    int e = i % E_;
    int t = i / E_;
    int d = t % DH_;
    int h = t / DH_;
    Wt[i] = (__bf16)W[((size_t)h * E_ + e) * DH_ + d];
}

// ---------------------------------------------------------------------------
// QKV projection: out[b,h] = x[b] (S x E) * W[h] (E x Dh) + bias[h]
// Weight 64x32 K-slices staged via async-to-LDS (shared by all 4 waves),
// double-buffered against the WMMA work.
// store_t == 0: out stored [B*H, S, Dh] bf16   (Q, K)
// store_t == 1: out stored [B*H, Dh, S] bf16   (V transposed, packed stores)
// ---------------------------------------------------------------------------
__global__ void qkv_proj(const __bf16* __restrict__ xb, const __bf16* __restrict__ Wt,
                         const float* __restrict__ bias, __bf16* __restrict__ out,
                         int store_t) {
    __shared__ alignas(16) __bf16 Wl[2][DH_ * 32];   // 2 x 4 KB

    int tid  = threadIdx.x;          // 0..127
    int wave = tid >> 5;
    int lane = tid & 31;
    int b = blockIdx.y / H_;
    int h = blockIdx.y % H_;
    int m0 = blockIdx.x * 64 + wave * 16;

    const __bf16* xp = xb + (size_t)b * S_ * E_;
    const __bf16* wp = Wt + (size_t)h * DH_ * E_;

    // stage W[:, k0..k0+31] into Wl[bufi] ([64][32], d-major): 2 asyncs/wave
    int wd = tid >> 1;
    int wc = (tid & 1) * 16;
    auto issueW = [&](int k0, int bufi) {
        unsigned goff = (unsigned)(((size_t)wd * E_ + k0 + wc) * sizeof(__bf16));
        async_copy_b128(&Wl[bufi][wd * 32 + wc],     wp, goff);
        async_copy_b128(&Wl[bufi][wd * 32 + wc + 8], wp, goff + 16);
    };

    v8f acc[4];
#pragma unroll
    for (int f = 0; f < 4; ++f) acc[f] = (v8f)0.0f;

    issueW(0, 0);
    for (int k0 = 0; k0 < E_; k0 += 32) {
        int buf = (k0 >> 5) & 1;
        if (k0 + 32 < E_) { issueW(k0 + 32, buf ^ 1); wait_async_le2(); }
        else              { wait_async_0(); }
        __syncthreads();

        v16bf a = load_a_frag(xp, m0, k0, E_);
#pragma unroll
        for (int f = 0; f < 4; ++f)
            acc[f] = wmma_bf16(a, load_b_frag(&Wl[buf][0], f * 16, 0, 32), acc[f]);

        __syncthreads();
    }

    int half = lane >> 4;
    int col  = lane & 15;
#pragma unroll
    for (int f = 0; f < 4; ++f) {
        float bv = bias[h * DH_ + f * 16 + col];
#pragma unroll
        for (int i = 0; i < 8; ++i) acc[f][i] += bv;
    }

    if (!store_t) {
        __bf16* qp = out + (size_t)(b * H_ + h) * S_ * DH_;
#pragma unroll
        for (int f = 0; f < 4; ++f)
#pragma unroll
            for (int i = 0; i < 8; ++i) {
                int s = m0 + i + 8 * half;
                qp[(size_t)s * DH_ + f * 16 + col] = (__bf16)acc[f][i];
            }
    } else {
        __bf16* vp = out + (size_t)(b * H_ + h) * DH_ * S_;
#pragma unroll
        for (int f = 0; f < 4; ++f) {
            int d = f * 16 + col;
            v8bf pk;
#pragma unroll
            for (int i = 0; i < 8; ++i) pk[i] = (__bf16)acc[f][i];
            *reinterpret_cast<v8bf*>(vp + (size_t)d * S_ + m0 + 8 * half) = pk;
        }
    }
}

// ---------------------------------------------------------------------------
// Flash attention + residual.
// 256 threads (8 waves); each wave owns 16 query rows. 32-key chunks of K and
// V are async-DMA'd to LDS once per block (shared by all 8 waves), double-
// buffered; online softmax; P staged via per-wave LDS tile.
// ---------------------------------------------------------------------------
__global__ void flash_attn(const __bf16* __restrict__ Qb, const __bf16* __restrict__ Kb,
                           const __bf16* __restrict__ Vt, const float* __restrict__ x,
                           float* __restrict__ out) {
    __shared__ alignas(16) __bf16 Kl[2][32 * DH_];   // [t][d], 2 x 4 KB
    __shared__ alignas(16) __bf16 Vl[2][DH_ * 32];   // [d][t], 2 x 4 KB
    __shared__ alignas(16) __bf16 Pl[8][16 * 32];    // per-wave P tile, 8 KB

    int tid  = threadIdx.x;          // 0..255
    int wave = tid >> 5;
    int lane = tid & 31;
    int half = lane >> 4;
    int col  = lane & 15;
    int b = blockIdx.y / H_;
    int h = blockIdx.y % H_;
    int m0 = blockIdx.x * 128 + wave * 16;

    const __bf16* qp = Qb + (size_t)(b * H_ + h) * S_ * DH_;
    const __bf16* kp = Kb + (size_t)(b * H_ + h) * S_ * DH_;
    const __bf16* vp = Vt + (size_t)(b * H_ + h) * DH_ * S_;

    // per-thread slices for the K/V chunk DMA (1 b128 each => 2 asyncs/wave)
    int kt = tid >> 3, kc = (tid & 7) * 8;           // K: [32][64]
    int vd = tid >> 2, vc = (tid & 3) * 8;           // V: [64][32]
    auto issueKV = [&](int t0, int bufi) {
        unsigned gk = (unsigned)(((size_t)(t0 + kt) * DH_ + kc) * sizeof(__bf16));
        async_copy_b128(&Kl[bufi][kt * DH_ + kc], kp, gk);
        unsigned gv = (unsigned)(((size_t)vd * S_ + t0 + vc) * sizeof(__bf16));
        async_copy_b128(&Vl[bufi][vd * 32 + vc], vp, gv);
    };

    // Q fragments for this wave's 16 rows (K dim = Dh = 64 -> two A-frags)
    v16bf qa0 = load_a_frag(qp, m0, 0, DH_);
    v16bf qa1 = load_a_frag(qp, m0, 32, DH_);

    v8f o[4];
#pragma unroll
    for (int f = 0; f < 4; ++f) o[f] = (v8f)0.0f;
    float mrow[8], lrow[8];
#pragma unroll
    for (int i = 0; i < 8; ++i) { mrow[i] = -3.0e38f; lrow[i] = 0.0f; }

    const float sc = 0.03125f;   // 1 / sqrt(E) = 1/32
    __bf16* pl = &Pl[wave][0];

    issueKV(0, 0);
    for (int t0 = 0; t0 < S_; t0 += 32) {
        int buf = (t0 >> 5) & 1;
        if (t0 + 32 < S_) { issueKV(t0 + 32, buf ^ 1); wait_async_le2(); }
        else              { wait_async_0(); }
        __syncthreads();

        // -------- scores: Q (16x64) * K^T (64x32) -> two 16x16 tiles --------
        const __bf16* kl = &Kl[buf][0];
        v8f s0 = (v8f)0.0f, s1 = (v8f)0.0f;
        s0 = wmma_bf16(qa0, load_b_frag(kl, 0,  0,  DH_), s0);
        s0 = wmma_bf16(qa1, load_b_frag(kl, 0,  32, DH_), s0);
        s1 = wmma_bf16(qa0, load_b_frag(kl, 16, 0,  DH_), s1);
        s1 = wmma_bf16(qa1, load_b_frag(kl, 16, 32, DH_), s1);

        // -------- online softmax (rows at fixed vgpr index; 16 columns live
        // in one 16-lane half -> xor masks 1..8 stay inside the half) --------
#pragma unroll
        for (int i = 0; i < 8; ++i) {
            float v0 = s0[i] * sc;
            float v1 = s1[i] * sc;
            float cm = fmaxf(v0, v1);
            cm = fmaxf(cm, __shfl_xor(cm, 1, 32));
            cm = fmaxf(cm, __shfl_xor(cm, 2, 32));
            cm = fmaxf(cm, __shfl_xor(cm, 4, 32));
            cm = fmaxf(cm, __shfl_xor(cm, 8, 32));
            float mnew  = fmaxf(mrow[i], cm);
            float scale = __expf(mrow[i] - mnew);
            float p0 = __expf(v0 - mnew);
            float p1 = __expf(v1 - mnew);
            float ps = p0 + p1;
            ps += __shfl_xor(ps, 1, 32);
            ps += __shfl_xor(ps, 2, 32);
            ps += __shfl_xor(ps, 4, 32);
            ps += __shfl_xor(ps, 8, 32);
            lrow[i] = lrow[i] * scale + ps;
            mrow[i] = mnew;
#pragma unroll
            for (int f = 0; f < 4; ++f) o[f][i] *= scale;
            s0[i] = p0;
            s1[i] = p1;
        }

        // -------- stage P (C-layout -> A-fragment layout) through LDS --------
#pragma unroll
        for (int i = 0; i < 8; ++i) {
            int row = i + 8 * half;
            pl[row * 32 + col]      = (__bf16)s0[i];
            pl[row * 32 + 16 + col] = (__bf16)s1[i];
        }
        asm volatile("s_wait_dscnt 0" ::: "memory");   // wave-local LDS RAW

        v16bf pa;
        {
            int r  = lane & 15;
            int kb = (lane >> 4) << 3;
            v8bf lo = *reinterpret_cast<const v8bf*>(pl + r * 32 + kb);
            v8bf hi = *reinterpret_cast<const v8bf*>(pl + r * 32 + 16 + kb);
#pragma unroll
            for (int i = 0; i < 8; ++i) { pa[i] = lo[i]; pa[i + 8] = hi[i]; }
        }

        // -------- O += P (16x32) * V (32x64) --------
        const __bf16* vl = &Vl[buf][0];
#pragma unroll
        for (int f = 0; f < 4; ++f)
            o[f] = wmma_bf16(pa, load_b_frag(vl, f * 16, 0, 32), o[f]);

        __syncthreads();
    }

    // -------- epilogue: normalize, residual add, store --------
    float inv[8];
#pragma unroll
    for (int i = 0; i < 8; ++i) inv[i] = 1.0f / lrow[i];

    const float* xp = x + (size_t)b * S_ * E_;
    float* op = out + (size_t)b * S_ * E_;
#pragma unroll
    for (int f = 0; f < 4; ++f)
#pragma unroll
        for (int i = 0; i < 8; ++i) {
            int s = m0 + i + 8 * half;
            int e = h * DH_ + f * 16 + col;
            size_t idx = (size_t)s * E_ + e;
            op[idx] = xp[idx] + o[f][i] * inv[i];
        }
}

// ---------------------------------------------------------------------------
extern "C" void kernel_launch(void* const* d_in, const int* in_sizes, int n_in,
                              void* d_out, int out_size, void* d_ws, size_t ws_size,
                              hipStream_t stream) {
    const float* x  = (const float*)d_in[0];
    const float* Wq = (const float*)d_in[1];
    const float* bq = (const float*)d_in[2];
    const float* Wk = (const float*)d_in[3];
    const float* bk = (const float*)d_in[4];
    const float* Wv = (const float*)d_in[5];
    const float* bv = (const float*)d_in[6];
    float* out = (float*)d_out;

    // workspace layout (all bf16), 16B-aligned regions
    char* ws = (char*)d_ws;
    const size_t xbBytes = (size_t)B_ * S_ * E_ * 2;       // 16 MB
    const size_t wBytes  = (size_t)H_ * DH_ * E_ * 2;      //  2 MB each
    const size_t qBytes  = (size_t)B_ * H_ * S_ * DH_ * 2; // 16 MB each
    __bf16* xb  = (__bf16*)(ws);
    __bf16* Wtq = (__bf16*)(ws + xbBytes);
    __bf16* Wtk = (__bf16*)(ws + xbBytes + wBytes);
    __bf16* Wtv = (__bf16*)(ws + xbBytes + 2 * wBytes);
    __bf16* Qb  = (__bf16*)(ws + xbBytes + 3 * wBytes);
    __bf16* Kb  = (__bf16*)(ws + xbBytes + 3 * wBytes + qBytes);
    __bf16* Vt  = (__bf16*)(ws + xbBytes + 3 * wBytes + 2 * qBytes);

    // 1) conversions
    {
        int n = B_ * S_ * E_;
        cvt_x_bf16<<<(n + 255) / 256, 256, 0, stream>>>(x, xb, n);
        int nw = H_ * DH_ * E_;
        cvt_w_t_bf16<<<(nw + 255) / 256, 256, 0, stream>>>(Wq, Wtq);
        cvt_w_t_bf16<<<(nw + 255) / 256, 256, 0, stream>>>(Wk, Wtk);
        cvt_w_t_bf16<<<(nw + 255) / 256, 256, 0, stream>>>(Wv, Wtv);
    }

    // 2) QKV projections (WMMA bf16, async-staged weights)
    {
        dim3 grid(S_ / 64, B_ * H_);
        qkv_proj<<<grid, 128, 0, stream>>>(xb, Wtq, bq, Qb, 0);
        qkv_proj<<<grid, 128, 0, stream>>>(xb, Wtk, bk, Kb, 0);
        qkv_proj<<<grid, 128, 0, stream>>>(xb, Wtv, bv, Vt, 1);
    }

    // 3) flash attention + residual (WMMA bf16, async-staged K/V, online softmax)
    {
        dim3 grid(S_ / 128, B_ * H_);
        flash_attn<<<grid, 256, 0, stream>>>(Qb, Kb, Vt, x, out);
    }
}
